// DAResBlock3D_61572651156139
// MI455X (gfx1250) — matrix operand order repaired and checked by reading, not compiled, not run on hardware
//
#include <hip/hip_runtime.h>

// ---------------------------------------------------------------------------
// CDNA5 (gfx1250) implementation of the dual-attention 3D res-block.
// All GEMM-shaped work (5 direct convs as implicit GEMM, q/k/v pointwise
// convs, PAM flash-attention, CAM channel attention) runs on
// v_wmma_f32_16x16x32_f16 (f16 inputs, f32 accumulate). BatchNorm/softmax
// stay in f32. PAM never materializes the 4096x4096 energy matrix.
// LDS tiles are stored transposed ([N][K]) so each lane's WMMA B-fragment is
// 32 contiguous, 16B-aligned bytes -> ds_load_b128 instead of ds_load_u16.
// ---------------------------------------------------------------------------

typedef __attribute__((ext_vector_type(16))) _Float16 v16h;
typedef __attribute__((ext_vector_type(8)))  float    v8f;

#define NBATCH 2
#define CCH    64
#define NVOX   4096          // 16*16*16
#define SLOPE  0.229166666667f
#define BN_EPS 1e-5f

// WMMA 16x16x32 f16 VGPR index maps used here (wave32; A/B K order is this code's, not the ISA's):
// A (16x32): lane r=lane&15 -> row M=r; half h=lane>>4;
//   element e -> K = h*16 + e
// B (32x16): lane r -> col N=r; element e -> K = h*16 + e   (32B contiguous)
// C/D (16x16 f32): lane r -> col N=r; element j -> row M = h*8 + j
__device__ __forceinline__ int kmapA(int e, int h) { return h * 16 + e; }
__device__ __forceinline__ int kmapB(int e, int h) { return h * 16 + e; }

__device__ __forceinline__ v8f wmma16(v16h a, v16h b, v8f c) {
  return __builtin_amdgcn_wmma_f32_16x16x32_f16(false, a, false, b, (short)0, c, false, false);
}

// ---------------------------------------------------------------------------
// f32 -> f16 pack
// ---------------------------------------------------------------------------
__global__ void pack_h_kernel(const float* __restrict__ x, _Float16* __restrict__ y, int n) {
  int i = blockIdx.x * blockDim.x + threadIdx.x;
  if (i < n) y[i] = (_Float16)x[i];
}

// ---------------------------------------------------------------------------
// Implicit-GEMM conv (SAME, stride 1). X:(B,CIN,4096) f16, W:(64,CIN*KS^3) f16.
// Block = 128 threads = 4 waves; wave w computes out-channel tile [16w,16w+16)
// for a 32-voxel tile (two accumulators -> weight A-fragment reused twice).
// K loop staged through LDS (im2col on the fly), tile stored [N][K] so the
// per-lane B-fragment read is two contiguous ds_load_b128.
// ---------------------------------------------------------------------------
template <int KS, int CIN>
__global__ void conv_wmma_kernel(const _Float16* __restrict__ X,
                                 const _Float16* __restrict__ W,
                                 const float* __restrict__ bias,
                                 float* __restrict__ Yf,
                                 _Float16* __restrict__ Yh,
                                 int co, int Ct) {
  constexpr int KV = KS * KS * KS;
  constexpr int KTOT = CIN * KV;
  const int b = blockIdx.x >> 7;
  const int nt = blockIdx.x & 127;      // 32-voxel tile index
  const int t = threadIdx.x;
  const int w = t >> 5;
  const int lane = t & 31;
  const int r = lane & 15;
  const int h = lane >> 4;
  const int obase = w * 16;

  // [n][k], rows padded to 40 halfs (80B) -> 16B-aligned lane chunks,
  // bank starts stride-4 across the 64 banks (conflict-free b128 reads).
  __shared__ _Float16 Bt[32][40];

  v8f acc0 = {}, acc1 = {};
  for (int kk = 0; kk < KTOT; kk += 32) {
    __syncthreads();
    // stage 32(K) x 32(N) im2col tile, transposed
#pragma unroll
    for (int i = 0; i < 8; ++i) {
      int e = t + 128 * i;               // 0..1023
      int kl = e >> 5, nl = e & 31;
      int kg = kk + kl;
      int ci = kg / KV, rk = kg % KV;
      int o0 = rk / (KS * KS) - KS / 2;
      int o1 = (rk / KS) % KS - KS / 2;
      int o2 = rk % KS - KS / 2;
      int n = nt * 32 + nl;
      int s0 = (n >> 8) + o0, s1 = ((n >> 4) & 15) + o1, s2 = (n & 15) + o2;
      _Float16 val = (_Float16)0.0f;
      if ((unsigned)s0 < 16u && (unsigned)s1 < 16u && (unsigned)s2 < 16u)
        val = X[((b * CIN + ci) << 12) + (s0 << 8) + (s1 << 4) + s2];
      Bt[nl][kl] = val;
    }
    __syncthreads();
    __builtin_prefetch(&W[(obase + r) * KTOT + kk + 32], 0, 1);
    v16h a, b0, b1;
#pragma unroll
    for (int e = 0; e < 16; ++e) {
      a[e] = W[(obase + r) * KTOT + kk + kmapA(e, h)];
      int kb = kmapB(e, h);
      b0[e] = Bt[r][kb];
      b1[e] = Bt[16 + r][kb];
    }
    acc0 = wmma16(a, b0, acc0);
    acc1 = wmma16(a, b1, acc1);
  }
#pragma unroll
  for (int j = 0; j < 8; ++j) {
    int o = obase + h * 8 + j;
    int n0 = nt * 32 + r;
    float bi = bias ? bias[o] : 0.0f;
    float y0 = acc0[j] + bi;
    float y1 = acc1[j] + bi;
    if (Yf) {
      Yf[((b * CCH + o) << 12) + n0] = y0;
      Yf[((b * CCH + o) << 12) + n0 + 16] = y1;
    }
    if (Yh) {
      Yh[((b * Ct + co + o) << 12) + n0] = (_Float16)y0;
      Yh[((b * Ct + co + o) << 12) + n0 + 16] = (_Float16)y1;
    }
  }
}

// ---------------------------------------------------------------------------
// BatchNorm batch statistics: one block per channel, reduce over (B, N).
// ---------------------------------------------------------------------------
__global__ void bn_stats_kernel(const float* __restrict__ x, float* __restrict__ mv) {
  const int c = blockIdx.x;
  const int t = threadIdx.x;
  float s = 0.f, s2 = 0.f;
  for (int b = 0; b < NBATCH; ++b)
    for (int i = t; i < NVOX; i += 256) {
      float v = x[((b * CCH + c) << 12) + i];
      s += v;
      s2 += v * v;
    }
  __shared__ float sh[256], sh2[256];
  sh[t] = s; sh2[t] = s2;
  __syncthreads();
  for (int off = 128; off > 0; off >>= 1) {
    if (t < off) { sh[t] += sh[t + off]; sh2[t] += sh2[t + off]; }
    __syncthreads();
  }
  if (t == 0) {
    const float inv = 1.0f / (NBATCH * NVOX);
    float m = sh[0] * inv;
    mv[c] = m;
    mv[CCH + c] = sh2[0] * inv - m * m;   // biased variance
  }
}

// ---------------------------------------------------------------------------
// Fused BN + RReLU(eval slope). Writes f32 (opt) and/or f16 copy (opt, with
// channel offset co into a Ct-channel tensor -> used for the concat buffer).
// ---------------------------------------------------------------------------
__global__ void bn_apply_kernel(const float* __restrict__ x, const float* __restrict__ mv,
                                const float* __restrict__ g, const float* __restrict__ bet,
                                float* __restrict__ yf, _Float16* __restrict__ yh,
                                int co, int Ct) {
  int idx = blockIdx.x * 256 + threadIdx.x;
  if (idx >= NBATCH * CCH * NVOX) return;
  int n = idx & (NVOX - 1);
  int c = (idx >> 12) & (CCH - 1);
  int b = idx >> 18;
  float m = mv[c], v = mv[CCH + c];
  float y = (x[idx] - m) * rsqrtf(v + BN_EPS) * g[c] + bet[c];
  y = (y >= 0.f) ? y : SLOPE * y;
  if (yf) yf[idx] = y;
  if (yh) yh[((b * Ct + co + c) << 12) + n] = (_Float16)y;
}

// ---------------------------------------------------------------------------
// PAM, flash-attention style. One wave per 16 query positions. Streams keys /
// values in 32-wide chunks, online softmax, P transposed through LDS for the
// second WMMA. Never materializes the 4096x4096 energy matrix.
// Pt is stored [nq][m] so the bp fragment read is contiguous b128s.
// ---------------------------------------------------------------------------
__global__ void pam_kernel(const _Float16* __restrict__ q, const _Float16* __restrict__ k,
                           const _Float16* __restrict__ v, const float* __restrict__ sIn,
                           const float* __restrict__ gamma, _Float16* __restrict__ out) {
  const int b = blockIdx.x >> 8;
  const int nt = blockIdx.x & 255;
  const int lane = threadIdx.x & 31;
  const int r = lane & 15;
  const int h = lane >> 4;

  __shared__ _Float16 Pt[16][40];       // [nq][m], padded rows (80B)
  __shared__ float corrS[16], lS[16];

  // A tiles: A[nq, c] = q[c, nq], K = c in two 32-chunks
  v16h aq0, aq1;
#pragma unroll
  for (int e = 0; e < 16; ++e) {
    int c0 = kmapA(e, h);
    int n = nt * 16 + r;
    aq0[e] = q[((b * CCH + c0) << 12) + n];
    aq1[e] = q[((b * CCH + 32 + c0) << 12) + n];
  }

  float mrun[8], lrun[8];
#pragma unroll
  for (int j = 0; j < 8; ++j) { mrun[j] = -3e38f; lrun[j] = 0.f; }
  v8f o0 = {}, o1 = {}, o2 = {}, o3 = {};

  for (int mb = 0; mb < NVOX; mb += 32) {
#pragma unroll
    for (int sub = 0; sub < 2; ++sub) {
      const int mbase = mb + sub * 16;
      // B tiles: B[c, m] = k[c, m]
      v16h bk0, bk1;
#pragma unroll
      for (int e = 0; e < 16; ++e) {
        int c0 = kmapB(e, h);
        bk0[e] = k[((b * CCH + c0) << 12) + mbase + r];
        bk1[e] = k[((b * CCH + 32 + c0) << 12) + mbase + r];
      }
      v8f et = {};
      et = wmma16(aq0, bk0, et);
      et = wmma16(aq1, bk1, et);

      // online softmax, rows nq = h*8+j spread over the 16-lane group
#pragma unroll
      for (int j = 0; j < 8; ++j) {
        float xv = et[j];
        float tm = xv;
        for (int off = 1; off < 16; off <<= 1) tm = fmaxf(tm, __shfl_xor(tm, off, 32));
        float nm = fmaxf(mrun[j], tm);
        float p = __expf(xv - nm);
        float ps = p;
        for (int off = 1; off < 16; off <<= 1) ps += __shfl_xor(ps, off, 32);
        float cf = __expf(mrun[j] - nm);
        lrun[j] = lrun[j] * cf + ps;
        mrun[j] = nm;
        if (r == 0) corrS[h * 8 + j] = cf;
        Pt[h * 8 + j][sub * 16 + r] = (_Float16)p;
      }
      __syncthreads();
      float cf = corrS[r];   // O cols are nq = r
#pragma unroll
      for (int j = 0; j < 8; ++j) { o0[j] *= cf; o1[j] *= cf; o2[j] *= cf; o3[j] *= cf; }
      __syncthreads();
    }
    // second GEMM: O[c, nq] += V[c, m(32)] * P^T[m, nq]
    v16h bp;
#pragma unroll
    for (int e = 0; e < 16; ++e) bp[e] = Pt[r][kmapB(e, h)];
#pragma unroll
    for (int ct = 0; ct < 4; ++ct) {
      v16h av;
#pragma unroll
      for (int e = 0; e < 16; ++e)
        av[e] = v[((b * CCH + ct * 16 + r) << 12) + mb + kmapA(e, h)];
      if (ct == 0) o0 = wmma16(av, bp, o0);
      else if (ct == 1) o1 = wmma16(av, bp, o1);
      else if (ct == 2) o2 = wmma16(av, bp, o2);
      else o3 = wmma16(av, bp, o3);
    }
    __syncthreads();
  }

  if (r == 0)
#pragma unroll
    for (int j = 0; j < 8; ++j) lS[h * 8 + j] = lrun[j];
  __syncthreads();
  const float li = 1.0f / lS[r];
  const float g = gamma[0];
#pragma unroll
  for (int ct = 0; ct < 4; ++ct)
#pragma unroll
    for (int j = 0; j < 8; ++j) {
      float ov = (ct == 0) ? o0[j] : (ct == 1) ? o1[j] : (ct == 2) ? o2[j] : o3[j];
      int c = ct * 16 + h * 8 + j;
      int n = nt * 16 + r;
      int idx = ((b * CCH + c) << 12) + n;
      out[idx] = (_Float16)(g * (ov * li) + sIn[idx]);
    }
}

// ---------------------------------------------------------------------------
// CAM energy: E[c,d] = <f[c,:], f[d,:]>, then softmax(rowmax - E) -> attn f16.
// One block (4 waves) per batch; K = 4096 streamed in 32-chunks.
// ---------------------------------------------------------------------------
__global__ void cam_energy_kernel(const _Float16* __restrict__ f, _Float16* __restrict__ attn) {
  const int b = blockIdx.x;
  const int t = threadIdx.x;
  const int w = t >> 5;
  const int lane = t & 31;
  const int r = lane & 15;
  const int h = lane >> 4;

  v8f acc0 = {}, acc1 = {}, acc2 = {}, acc3 = {};
  for (int nb = 0; nb < NVOX; nb += 32) {
    v16h a;
#pragma unroll
    for (int e = 0; e < 16; ++e)
      a[e] = f[((b * CCH + w * 16 + r) << 12) + nb + kmapA(e, h)];
#pragma unroll
    for (int dt = 0; dt < 4; ++dt) {
      v16h bb;
#pragma unroll
      for (int e = 0; e < 16; ++e)
        bb[e] = f[((b * CCH + dt * 16 + r) << 12) + nb + kmapB(e, h)];
      if (dt == 0) acc0 = wmma16(a, bb, acc0);
      else if (dt == 1) acc1 = wmma16(a, bb, acc1);
      else if (dt == 2) acc2 = wmma16(a, bb, acc2);
      else acc3 = wmma16(a, bb, acc3);
    }
  }
  __shared__ float E[64][65];
#pragma unroll
  for (int dt = 0; dt < 4; ++dt)
#pragma unroll
    for (int j = 0; j < 8; ++j) {
      float vv = (dt == 0) ? acc0[j] : (dt == 1) ? acc1[j] : (dt == 2) ? acc2[j] : acc3[j];
      E[w * 16 + h * 8 + j][dt * 16 + r] = vv;
    }
  __syncthreads();
  if (t < 64) {
    int c = t;
    float rmax = -3e38f, rmin = 3e38f;
    for (int d = 0; d < 64; ++d) { rmax = fmaxf(rmax, E[c][d]); rmin = fminf(rmin, E[c][d]); }
    float enmax = rmax - rmin;     // max of (rmax - E)
    float s = 0.f;
    for (int d = 0; d < 64; ++d) s += __expf((rmax - E[c][d]) - enmax);
    float inv = 1.0f / s;
    for (int d = 0; d < 64; ++d)
      attn[(b * CCH + c) * 64 + d] = (_Float16)(__expf((rmax - E[c][d]) - enmax) * inv);
  }
}

// ---------------------------------------------------------------------------
// CAM apply: out[c,n] = gamma * sum_d attn[c,d] f[d,n] + x[c,n], f16 out.
// ---------------------------------------------------------------------------
__global__ void cam_apply_kernel(const _Float16* __restrict__ attn, const _Float16* __restrict__ f,
                                 const float* __restrict__ xres, const float* __restrict__ gamma,
                                 _Float16* __restrict__ out) {
  const int b = blockIdx.x >> 8;
  const int nt = blockIdx.x & 255;
  const int t = threadIdx.x;
  const int w = t >> 5;
  const int lane = t & 31;
  const int r = lane & 15;
  const int h = lane >> 4;

  v16h a0, a1, b0, b1;
#pragma unroll
  for (int e = 0; e < 16; ++e) {
    a0[e] = attn[(b * CCH + w * 16 + r) * 64 + kmapA(e, h)];
    a1[e] = attn[(b * CCH + w * 16 + r) * 64 + 32 + kmapA(e, h)];
    int d = kmapB(e, h);
    b0[e] = f[((b * CCH + d) << 12) + nt * 16 + r];
    b1[e] = f[((b * CCH + 32 + d) << 12) + nt * 16 + r];
  }
  v8f acc = {};
  acc = wmma16(a0, b0, acc);
  acc = wmma16(a1, b1, acc);
  const float g = gamma[0];
#pragma unroll
  for (int j = 0; j < 8; ++j) {
    int c = w * 16 + h * 8 + j;
    int n = nt * 16 + r;
    int idx = ((b * CCH + c) << 12) + n;
    out[idx] = (_Float16)(g * acc[j] + xres[idx]);
  }
}

// ---------------------------------------------------------------------------
// Host orchestration
// ---------------------------------------------------------------------------
static inline void pack_h(const float* src, _Float16* dst, int n, hipStream_t s) {
  pack_h_kernel<<<(n + 255) / 256, 256, 0, s>>>(src, dst, n);
}

extern "C" void kernel_launch(void* const* d_in, const int* in_sizes, int n_in,
                              void* d_out, int out_size, void* d_ws, size_t ws_size,
                              hipStream_t stream) {
  const float* x   = (const float*)d_in[0];
  const float* wS  = (const float*)d_in[1];
  const float* gS  = (const float*)d_in[2];
  const float* bS  = (const float*)d_in[3];
  const float* qw  = (const float*)d_in[4];
  const float* qb  = (const float*)d_in[5];
  const float* kw  = (const float*)d_in[6];
  const float* kb  = (const float*)d_in[7];
  const float* vw  = (const float*)d_in[8];
  const float* vb  = (const float*)d_in[9];
  const float* gmp = (const float*)d_in[10];
  const float* wS1 = (const float*)d_in[11];
  const float* gS1 = (const float*)d_in[12];
  const float* bS1 = (const float*)d_in[13];
  const float* wC  = (const float*)d_in[14];
  const float* gC  = (const float*)d_in[15];
  const float* bC  = (const float*)d_in[16];
  const float* gmc = (const float*)d_in[17];
  const float* wC1 = (const float*)d_in[18];
  const float* gC1 = (const float*)d_in[19];
  const float* bC1 = (const float*)d_in[20];
  const float* wF  = (const float*)d_in[21];
  const float* gF  = (const float*)d_in[22];
  const float* bF  = (const float*)d_in[23];

  const int ACT = NBATCH * CCH * NVOX;      // 524288
  const int W3  = CCH * CCH * 27;           // 110592
  const int WF  = CCH * 2 * CCH * 27;       // 221184
  const int W1  = CCH * CCH;                // 4096

  // carve workspace
  char* p = (char*)d_ws;
  auto take = [&](size_t bytes) -> char* {
    char* r = p;
    p += (bytes + 255) & ~(size_t)255;
    return r;
  };
  float* conv_tmp = (float*)take((size_t)ACT * 4);
  float* sS_f32   = (float*)take((size_t)ACT * 4);
  float* cC_f32   = (float*)take((size_t)ACT * 4);
  float* mv       = (float*)take(2 * CCH * 4);
  _Float16* x_h    = (_Float16*)take((size_t)ACT * 2);
  _Float16* sS_h   = (_Float16*)take((size_t)ACT * 2);
  _Float16* q_h    = (_Float16*)take((size_t)ACT * 2);
  _Float16* k_h    = (_Float16*)take((size_t)ACT * 2);
  _Float16* v_h    = (_Float16*)take((size_t)ACT * 2);
  _Float16* pam_h  = (_Float16*)take((size_t)ACT * 2);
  _Float16* cC_h   = (_Float16*)take((size_t)ACT * 2);
  _Float16* cam_h  = (_Float16*)take((size_t)ACT * 2);
  _Float16* cat_h  = (_Float16*)take((size_t)ACT * 2 * 2);  // (B, 2C, N)
  _Float16* wS_h   = (_Float16*)take((size_t)W3 * 2);
  _Float16* wS1_h  = (_Float16*)take((size_t)W3 * 2);
  _Float16* wC_h   = (_Float16*)take((size_t)W3 * 2);
  _Float16* wC1_h  = (_Float16*)take((size_t)W3 * 2);
  _Float16* wF_h   = (_Float16*)take((size_t)WF * 2);
  _Float16* qw_h   = (_Float16*)take((size_t)W1 * 2);
  _Float16* kw_h   = (_Float16*)take((size_t)W1 * 2);
  _Float16* vw_h   = (_Float16*)take((size_t)W1 * 2);
  _Float16* attn_h = (_Float16*)take((size_t)NBATCH * CCH * CCH * 2);

  // pack inputs + weights to f16
  pack_h(x, x_h, ACT, stream);
  pack_h(wS, wS_h, W3, stream);
  pack_h(wS1, wS1_h, W3, stream);
  pack_h(wC, wC_h, W3, stream);
  pack_h(wC1, wC1_h, W3, stream);
  pack_h(wF, wF_h, WF, stream);
  pack_h(qw, qw_h, W1, stream);
  pack_h(kw, kw_h, W1, stream);
  pack_h(vw, vw_h, W1, stream);

  const dim3 cgrid(NBATCH * 128), cblk(128);
  const dim3 egrid((ACT + 255) / 256), eblk(256);

  // ---- spatial branch ----
  conv_wmma_kernel<3, 64><<<cgrid, cblk, 0, stream>>>(x_h, wS_h, nullptr, conv_tmp, nullptr, 0, CCH);
  bn_stats_kernel<<<CCH, 256, 0, stream>>>(conv_tmp, mv);
  bn_apply_kernel<<<egrid, eblk, 0, stream>>>(conv_tmp, mv, gS, bS, sS_f32, sS_h, 0, CCH);

  conv_wmma_kernel<1, 64><<<cgrid, cblk, 0, stream>>>(sS_h, qw_h, qb, nullptr, q_h, 0, CCH);
  conv_wmma_kernel<1, 64><<<cgrid, cblk, 0, stream>>>(sS_h, kw_h, kb, nullptr, k_h, 0, CCH);
  conv_wmma_kernel<1, 64><<<cgrid, cblk, 0, stream>>>(sS_h, vw_h, vb, nullptr, v_h, 0, CCH);

  pam_kernel<<<NBATCH * 256, 32, 0, stream>>>(q_h, k_h, v_h, sS_f32, gmp, pam_h);

  conv_wmma_kernel<3, 64><<<cgrid, cblk, 0, stream>>>(pam_h, wS1_h, nullptr, conv_tmp, nullptr, 0, CCH);
  bn_stats_kernel<<<CCH, 256, 0, stream>>>(conv_tmp, mv);
  bn_apply_kernel<<<egrid, eblk, 0, stream>>>(conv_tmp, mv, gS1, bS1, nullptr, cat_h, 0, 2 * CCH);

  // ---- channel branch ----
  conv_wmma_kernel<3, 64><<<cgrid, cblk, 0, stream>>>(x_h, wC_h, nullptr, conv_tmp, nullptr, 0, CCH);
  bn_stats_kernel<<<CCH, 256, 0, stream>>>(conv_tmp, mv);
  bn_apply_kernel<<<egrid, eblk, 0, stream>>>(conv_tmp, mv, gC, bC, cC_f32, cC_h, 0, CCH);

  cam_energy_kernel<<<NBATCH, 128, 0, stream>>>(cC_h, attn_h);
  cam_apply_kernel<<<NBATCH * 256, 128, 0, stream>>>(attn_h, cC_h, cC_f32, gmc, cam_h);

  conv_wmma_kernel<3, 64><<<cgrid, cblk, 0, stream>>>(cam_h, wC1_h, nullptr, conv_tmp, nullptr, 0, CCH);
  bn_stats_kernel<<<CCH, 256, 0, stream>>>(conv_tmp, mv);
  bn_apply_kernel<<<egrid, eblk, 0, stream>>>(conv_tmp, mv, gC1, bC1, nullptr, cat_h, CCH, 2 * CCH);

  // ---- fuse ----
  conv_wmma_kernel<3, 128><<<cgrid, cblk, 0, stream>>>(cat_h, wF_h, nullptr, conv_tmp, nullptr, 0, CCH);
  bn_stats_kernel<<<CCH, 256, 0, stream>>>(conv_tmp, mv);
  bn_apply_kernel<<<egrid, eblk, 0, stream>>>(conv_tmp, mv, gF, bF, (float*)d_out, nullptr, 0, CCH);
}
